// NodeAttention_16758962389077
// MI455X (gfx1250) — matrix operand order repaired and checked by reading, not compiled, not run on hardware
//
#include <hip/hip_runtime.h>
#include <hip/hip_bf16.h>
#include <stdint.h>

#define N_NODES 8192
#define D_FEAT  128

typedef __attribute__((ext_vector_type(16))) __bf16 v16bf;
typedef __attribute__((ext_vector_type(8)))  float  v8f;

static __device__ __forceinline__ unsigned short f2bf(float f) {
    // round-to-nearest-even f32 -> bf16
    uint32_t u = __float_as_uint(f);
    uint32_t r = (u + 0x7FFFu + ((u >> 16) & 1u)) >> 16;
    return (unsigned short)r;
}

union APack { uint32_t u[8]; v16bf v; };
union BPack { uint4 q[2];    v16bf v; };

// ---------------------------------------------------------------------------
// prep: s[j] = emb[j,:] . H_v   and   emb -> bf16 (for WMMA B operand)
// ---------------------------------------------------------------------------
__global__ __launch_bounds__(128) void prep_kernel(
    const float* __restrict__ emb, const float* __restrict__ Hv,
    float* __restrict__ s, unsigned short* __restrict__ embbf)
{
    const int j = blockIdx.x;
    const int d = threadIdx.x;
    float v = emb[(size_t)j * D_FEAT + d];
    embbf[(size_t)j * D_FEAT + d] = f2bf(v);

    __shared__ float red[128];
    red[d] = v * Hv[d];
    __syncthreads();
    for (int off = 64; off > 0; off >>= 1) {
        if (d < off) red[d] += red[d + off];
        __syncthreads();
    }
    if (d == 0) s[j] = red[0];
}

// ---------------------------------------------------------------------------
// attn: per 16-row stripe: online masked softmax stats, then fused
// p-staging (bf16) + v_wmma_f32_16x16x32_bf16 GEMM against emb_bf16.
// Block = 256 threads = 8 waves; wave w owns output columns [16w, 16w+16).
// ---------------------------------------------------------------------------
__global__ __launch_bounds__(256) void attn_kernel(
    const float* __restrict__ adj, const float* __restrict__ s,
    const unsigned short* __restrict__ embbf, float* __restrict__ out)
{
    const int i0 = blockIdx.x * 16;
    const int t  = threadIdx.x;
    const int r  = t >> 4;     // row within 16-row stripe (16 threads/row)
    const int c0 = t & 15;

    __shared__ float red_m[16][16];
    __shared__ float red_s[16][16];
    __shared__ float row_m[16];
    __shared__ float row_rz[16];
    __shared__ unsigned short pstage[16][258];  // 256 + 2 pad (bank spread)

    const size_t rowoff = (size_t)(i0 + r) * N_NODES;

    // ---- Pass A: online (max, sum-exp) over nonzero entries of the row ----
    float mx = -3.0e38f, sm = 0.0f;
    for (int j = c0; j < N_NODES; j += 16) {
        float a = adj[rowoff + j];
        if (a != 0.0f) {
            float l = a * s[j];
            if (l > mx) { sm = sm * __expf(mx - l) + 1.0f; mx = l; }
            else        { sm += __expf(l - mx); }
        }
    }
    red_m[r][c0] = mx;
    red_s[r][c0] = sm;
    __syncthreads();
    if (t < 16) {
        float M = -3.0e38f, S = 0.0f;
        for (int c = 0; c < 16; ++c) {
            float m2 = red_m[t][c], s2 = red_s[t][c];
            if (m2 > M) { S = S * __expf(M - m2) + s2; M = m2; }
            else        { S += s2 * __expf(m2 - M); }
        }
        row_m[t]  = M;
        row_rz[t] = 1.0f / S;   // S >= 1 guaranteed by self-loops
    }
    __syncthreads();

    // ---- Pass B: fused alpha staging + WMMA accumulation over K=8192 ----
    const int wave  = t >> 5;          // 0..7
    const int lane  = t & 31;
    const int d0    = wave * 16;       // output column base for this wave
    const int m     = lane & 15;       // A-matrix row held by this lane
    const int kbase = (lane & 16) ? 8 : 0;  // A K-offset for upper lane half

    const float Mr  = row_m[r];
    const float RZr = row_rz[r];

    v8f acc = {};

    for (int chunk = 0; chunk < N_NODES; chunk += 256) {
        // stage p = exp(l - m)/Z as bf16 for this 16 x 256 alpha tile
        #pragma unroll 4
        for (int q = 0; q < 16; ++q) {
            int j = chunk + c0 + q * 16;
            float a = adj[rowoff + j];
            float p = 0.0f;
            if (a != 0.0f) p = __expf(a * s[j] - Mr) * RZr;
            pstage[r][c0 + q * 16] = f2bf(p);
        }
        __syncthreads();

        #pragma unroll
        for (int kk = 0; kk < 8; ++kk) {
            // A: 16x32 bf16 tile from LDS (ISA per-lane layout):
            //   lane m holds M=m; K pairs {0,2,4,6 | 16,18,20,22} (+8 upper half)
            APack A;
            const uint32_t* prow =
                ((const uint32_t*)&pstage[m][0]) + kk * 16 + (kbase >> 1);
            A.u[0] = prow[0];  A.u[1] = prow[1];
            A.u[2] = prow[2];  A.u[3] = prow[3];
            A.u[4] = prow[8];  A.u[5] = prow[9];
            A.u[6] = prow[10]; A.u[7] = prow[11];

            // B: 32x16 bf16 tile; N = d0 + (lane & 15), K in same order as A
            BPack B;
            #pragma unroll
            for (int e = 0; e < 16; ++e) {
                const int kg = chunk + kk * 32 + kbase + (e < 8 ? e : 8 + e);
                ((unsigned short*)&B)[e] =
                    embbf[(size_t)kg * D_FEAT + d0 + (lane & 15)];
            }

            acc = __builtin_amdgcn_wmma_f32_16x16x32_bf16(
                false, A.v, false, B.v, (short)0, acc, false, false);
        }
        __syncthreads();
    }

    // ---- store C (f32 16x16 layout: VGPR rr -> M=rr / rr+8 by lane half) ----
    const int cm = (lane & 16) ? 8 : 0;
    const int cn = lane & 15;
    #pragma unroll
    for (int rr = 0; rr < 8; ++rr) {
        out[(size_t)(i0 + rr + cm) * D_FEAT + d0 + cn] = acc[rr];
    }
}

// ---------------------------------------------------------------------------
extern "C" void kernel_launch(void* const* d_in, const int* in_sizes, int n_in,
                              void* d_out, int out_size, void* d_ws, size_t ws_size,
                              hipStream_t stream) {
    (void)in_sizes; (void)n_in; (void)out_size; (void)ws_size;
    const float* emb = (const float*)d_in[0];   // [N, D] f32
    const float* adj = (const float*)d_in[1];   // [N, N] f32 (dense-stored sparse)
    const float* Hv  = (const float*)d_in[2];   // [D, 1] f32
    float* out = (float*)d_out;                 // [N, D] f32

    float* s_ws = (float*)d_ws;                                  // N floats
    unsigned short* embbf =
        (unsigned short*)((char*)d_ws + N_NODES * sizeof(float)); // N*D bf16

    prep_kernel<<<N_NODES, D_FEAT, 0, stream>>>(emb, Hv, s_ws, embbf);
    attn_kernel<<<N_NODES / 16, 256, 0, stream>>>(adj, s_ws, embbf, out);
}
